// QecTransformer_3066606649716
// MI455X (gfx1250) — matrix operand order repaired and checked by reading, not compiled, not run on hardware
//
#include <hip/hip_runtime.h>
#include <hip/hip_bf16.h>

// ---------------- constants (match reference shapes) ----------------
constexpr int kB   = 64;
constexpr int kS   = 440;   // tokens per batch
constexpr int kST  = 448;   // padded to 28 tiles of 16
constexpr int kD   = 256;
constexpr int kH   = 8;
constexpr int kDH  = 32;
constexpr int kF   = 512;
constexpr int kL   = 6;
constexpr int kG   = 22;    // grid side
constexpr int kNTOK = kB * kS;          // 28160 (multiple of 32)
constexpr int kMT32 = kNTOK / 32;       // 880 M-tiles (32 rows each)

typedef __attribute__((ext_vector_type(16))) __bf16 v16bf;
typedef __attribute__((ext_vector_type(8)))  __bf16 v8bf;
typedef __attribute__((ext_vector_type(8)))  float  v8f;

__device__ __forceinline__ v16bf combine16(v8bf lo, v8bf hi) {
  return __builtin_shufflevector(lo, hi, 0,1,2,3,4,5,6,7,8,9,10,11,12,13,14,15);
}
__device__ __forceinline__ v8f wmma_bf16(v16bf a, v16bf b, v8f c) {
  return __builtin_amdgcn_wmma_f32_16x16x32_bf16(false, a, false, b, (short)0, c, false, false);
}

// ---------------- f32 -> bf16 weight convert ----------------
__global__ __launch_bounds__(256) void cvt_kernel(const float* __restrict__ src,
                                                  __bf16* __restrict__ dst, long n) {
  long i = (long)blockIdx.x * 256 + threadIdx.x;
  if (i < n) dst[i] = (__bf16)src[i];
}

// ---------------- embedding: h = event_emb[x] + pos_x + pos_y + stab ----------------
__global__ __launch_bounds__(256) void embed_kernel(
    const int* __restrict__ x, const float* __restrict__ ee,
    const float* __restrict__ px, const float* __restrict__ py,
    const float* __restrict__ se, const int* __restrict__ tx,
    const int* __restrict__ ty, const int* __restrict__ ts,
    float* __restrict__ h, __bf16* __restrict__ hbf) {
  long idx = (long)blockIdx.x * 256 + threadIdx.x;   // NTOK*D threads
  int d = (int)(idx & (kD - 1));
  long tok = idx >> 8;
  int s = (int)(tok % kS);
  int b = (int)(tok / kS);
  float v = ee[(size_t)x[(size_t)b*kS + s]*kD + d]
          + px[(size_t)tx[s]*kD + d]
          + py[(size_t)ty[s]*kD + d]
          + se[(size_t)ts[s]*kD + d];
  h[idx] = v;
  hbf[idx] = (__bf16)v;
}

// ---------------- bf16 GEMM: C(MxN) = A(MxK) * W(NxK)^T + bias ----------------
// one wave computes a 32x64 tile of C (8 WMMA accumulators), K software-pipelined
__global__ __launch_bounds__(32) void gemm_bf16_kernel(
    const __bf16* __restrict__ A, const __bf16* __restrict__ W,
    const float* __restrict__ bias, float* __restrict__ outF,
    __bf16* __restrict__ outB, int N, int K, int relu) {
  const int lane = threadIdx.x;
  const int lm = lane & 15, hf = lane >> 4;
  const int mt = blockIdx.x;   // 32 rows
  const int nt4 = blockIdx.y;  // 64 cols
  const __bf16* arow0 = A + (size_t)(mt * 32 + lm) * K;
  const __bf16* arow1 = arow0 + (size_t)16 * K;
  const __bf16* wrow  = W + (size_t)(nt4 * 64 + lm) * K;

  auto ldA = [&](const __bf16* p, int k0) -> v16bf {
    return combine16(*(const v8bf*)(p + k0 + hf * 16),
                     *(const v8bf*)(p + k0 + hf * 16 + 8));
  };
  auto ldB = [&](int j, int k0) -> v16bf {
    const __bf16* wp = wrow + (size_t)j * 16 * K + k0 + hf * 16;
    return combine16(*(const v8bf*)wp, *(const v8bf*)(wp + 8));
  };

  v8f acc[2][4];
#pragma unroll
  for (int i = 0; i < 2; ++i)
#pragma unroll
    for (int j = 0; j < 4; ++j) acc[i][j] = v8f{};

  // prologue: stage K-step 0
  v16bf a0 = ldA(arow0, 0), a1 = ldA(arow1, 0);
  v16bf b0 = ldB(0, 0), b1 = ldB(1, 0), b2 = ldB(2, 0), b3 = ldB(3, 0);

  const int steps = K >> 5;
  for (int st = 0; st < steps - 1; ++st) {
    const int kn = (st + 1) << 5;
    // prefetch next K-step while current WMMAs execute
    v16bf na0 = ldA(arow0, kn), na1 = ldA(arow1, kn);
    v16bf nb0 = ldB(0, kn), nb1 = ldB(1, kn), nb2 = ldB(2, kn), nb3 = ldB(3, kn);
    acc[0][0] = wmma_bf16(a0, b0, acc[0][0]);
    acc[1][0] = wmma_bf16(a1, b0, acc[1][0]);
    acc[0][1] = wmma_bf16(a0, b1, acc[0][1]);
    acc[1][1] = wmma_bf16(a1, b1, acc[1][1]);
    acc[0][2] = wmma_bf16(a0, b2, acc[0][2]);
    acc[1][2] = wmma_bf16(a1, b2, acc[1][2]);
    acc[0][3] = wmma_bf16(a0, b3, acc[0][3]);
    acc[1][3] = wmma_bf16(a1, b3, acc[1][3]);
    a0 = na0; a1 = na1; b0 = nb0; b1 = nb1; b2 = nb2; b3 = nb3;
  }
  acc[0][0] = wmma_bf16(a0, b0, acc[0][0]);
  acc[1][0] = wmma_bf16(a1, b0, acc[1][0]);
  acc[0][1] = wmma_bf16(a0, b1, acc[0][1]);
  acc[1][1] = wmma_bf16(a1, b1, acc[1][1]);
  acc[0][2] = wmma_bf16(a0, b2, acc[0][2]);
  acc[1][2] = wmma_bf16(a1, b2, acc[1][2]);
  acc[0][3] = wmma_bf16(a0, b3, acc[0][3]);
  acc[1][3] = wmma_bf16(a1, b3, acc[1][3]);

#pragma unroll
  for (int i = 0; i < 2; ++i) {
    const int rowb = mt * 32 + i * 16 + 8 * hf;
#pragma unroll
    for (int r = 0; r < 8; ++r) {
      size_t ro = (size_t)(rowb + r) * N;
#pragma unroll
      for (int j = 0; j < 4; ++j) {
        int col = nt4 * 64 + j * 16 + lm;
        float v = acc[i][j][r] + bias[col];
        if (relu) v = fmaxf(v, 0.f);
        if (outF) outF[ro + col] = v;
        if (outB) outB[ro + col] = (__bf16)v;
      }
    }
  }
}

// ---------------- fused attention: one wave per (16-query tile, b, head) ----------------
__global__ __launch_bounds__(32) void attn_kernel(
    const __bf16* __restrict__ qkv, const float* __restrict__ abias,
    __bf16* __restrict__ o) {
  __shared__ __align__(16) float  sc[16 * kST];      // scores f32
  __shared__ __align__(16) __bf16 pP[16 * kST];      // softmax probs bf16
  __shared__ __align__(16) __bf16 vT[kDH * 224];     // V^T, half the keys at a time
  __shared__ float redmx[2][16];
  __shared__ float redsm[2][16];

  const float scale = 0.17677669529663687f;          // 1/sqrt(32)
  const int lane = threadIdx.x;
  const int lm = lane & 15, hf = lane >> 4;
  const int qt = blockIdx.x;                         // 0..27
  const int bh = blockIdx.y;                         // 0..511
  const int b = bh >> 3, hh = bh & 7;
  const size_t rs = 3 * kD;
  const __bf16* qb = qkv + (size_t)b * kS * rs + hh * kDH;
  const __bf16* kb = qb + kD;
  const __bf16* vb = qb + 2 * kD;

  // A = q tile (16 x 32), held for whole kernel
  int qrow = qt * 16 + lm;
  int qr = qrow < kS ? qrow : kS - 1;
  const __bf16* qp = qb + (size_t)qr * rs;
  v16bf amat = combine16(*(const v8bf*)(qp + hf * 16),
                         *(const v8bf*)(qp + hf * 16 + 8));

  auto ldK = [&](int kt) -> v16bf {
    int krow = kt * 16 + lm;
    int kr = krow < kS ? krow : kS - 1;
    const __bf16* kp = kb + (size_t)kr * rs + hf * 16;
    return combine16(*(const v8bf*)kp, *(const v8bf*)(kp + 8));
  };

  // ---- scores = (q k^T)*scale + bias, key tiles pipelined ----
  v16bf bmat = ldK(0);
  for (int kt = 0; kt < 28; ++kt) {
    v16bf bnext = (kt < 27) ? ldK(kt + 1) : bmat;
    v8f c = v8f{};
    c = wmma_bf16(amat, bmat, c);
    int krow = kt * 16 + lm;
#pragma unroll
    for (int r = 0; r < 8; ++r) {
      int m = r + 8 * hf;
      float v;
      if (krow < kS) {
        int qq = qt * 16 + m; if (qq >= kS) qq = kS - 1;
        v = c[r] * scale + abias[(size_t)qq * kS + krow];
      } else {
        v = -3.0e38f;                                // mask padded keys
      }
      sc[m * kST + krow] = v;
    }
    bmat = bnext;
  }
  __syncthreads();

  // ---- row softmax (lane pair (lm, lm+16) splits each row) ----
  {
    float* rowp = sc + lm * kST + hf * 224;
    float mx = -3.4e38f;
    for (int j = 0; j < 224; ++j) mx = fmaxf(mx, rowp[j]);
    redmx[hf][lm] = mx;
    __syncthreads();
    mx = fmaxf(redmx[0][lm], redmx[1][lm]);
    float sum = 0.f;
    for (int j = 0; j < 224; ++j) { float e = __expf(rowp[j] - mx); rowp[j] = e; sum += e; }
    redsm[hf][lm] = sum;
    __syncthreads();
    float inv = 1.f / (redsm[0][lm] + redsm[1][lm]);
    __bf16* prow = pP + lm * kST + hf * 224;
    for (int j = 0; j < 224; ++j) prow[j] = (__bf16)(rowp[j] * inv);
  }
  __syncthreads();

  // ---- o = P V  (two key halves; V^T staged in LDS) ----
  v8f o0 = v8f{}, o1 = v8f{};
  for (int hs = 0; hs < 2; ++hs) {
    for (int idx = lane; idx < 224 * 16; idx += 32) {
      int key = idx >> 4;
      int cp = (idx & 15) << 1;
      int gk = hs * 224 + key;
      int kr2 = gk < kS ? gk : kS - 1;
      const __bf16* vp = vb + (size_t)kr2 * rs + cp;
      vT[cp * 224 + key]       = vp[0];
      vT[(cp + 1) * 224 + key] = vp[1];
    }
    __syncthreads();
    for (int kt2 = 0; kt2 < 7; ++kt2) {
      const __bf16* pr = pP + lm * kST + hs * 224 + kt2 * 32;
      v16bf ap = combine16(*(const v8bf*)(pr + hf * 16),
                           *(const v8bf*)(pr + hf * 16 + 8));
      const __bf16* v0 = vT + lm * 224 + kt2 * 32 + hf * 16;
      v16bf pb0 = combine16(*(const v8bf*)v0, *(const v8bf*)(v0 + 8));
      o0 = wmma_bf16(ap, pb0, o0);
      const __bf16* v1 = vT + (16 + lm) * 224 + kt2 * 32 + hf * 16;
      v16bf pb1 = combine16(*(const v8bf*)v1, *(const v8bf*)(v1 + 8));
      o1 = wmma_bf16(ap, pb1, o1);
    }
    __syncthreads();
  }

#pragma unroll
  for (int r = 0; r < 8; ++r) {
    int qq = qt * 16 + r + 8 * hf;
    if (qq < kS) {
      size_t ob = ((size_t)b * kS + qq) * kD + hh * kDH;
      o[ob + lm]      = (__bf16)o0[r];
      o[ob + 16 + lm] = (__bf16)o1[r];
    }
  }
}

// ---------------- layernorm of (h + res), in-place on h ----------------
__global__ __launch_bounds__(256) void ln_kernel(
    float* __restrict__ h, const float* __restrict__ res,
    const float* __restrict__ gamma, const float* __restrict__ beta,
    __bf16* __restrict__ hbf) {
  __shared__ float sb[2][8];
  __shared__ float mv[2];
  const long tok = blockIdx.x;
  const int t = threadIdx.x;
  const size_t i = (size_t)tok * kD + t;
  float v = h[i] + res[i];
  float s1 = v, s2 = v * v;
#pragma unroll
  for (int off = 16; off > 0; off >>= 1) {
    s1 += __shfl_down(s1, off, 32);
    s2 += __shfl_down(s2, off, 32);
  }
  const int wid = t >> 5, lid = t & 31;
  if (lid == 0) { sb[0][wid] = s1; sb[1][wid] = s2; }
  __syncthreads();
  if (t == 0) {
    float a = 0.f, q = 0.f;
    for (int w = 0; w < 8; ++w) { a += sb[0][w]; q += sb[1][w]; }
    float m = a * (1.f / kD);
    float var = q * (1.f / kD) - m * m;
    mv[0] = m; mv[1] = rsqrtf(var + 1e-5f);
  }
  __syncthreads();
  float outv = (v - mv[0]) * mv[1] * gamma[t] + beta[t];
  h[i] = outv;
  if (hbf) hbf[i] = (__bf16)outv;
}

// ---------------- zero + scatter to grid ----------------
__global__ __launch_bounds__(256) void zero_kernel(float* __restrict__ p, long n) {
  long i = (long)blockIdx.x * 256 + threadIdx.x;
  long st = (long)gridDim.x * 256;
  for (; i < n; i += st) p[i] = 0.f;
}

__global__ __launch_bounds__(256) void scatter_kernel(
    const float* __restrict__ h, const int* __restrict__ spos,
    float* __restrict__ gridb) {
  long idx = (long)blockIdx.x * 256 + threadIdx.x;    // NTOK*D
  int d = (int)(idx & (kD - 1));
  long tok = idx >> 8;
  int s = (int)(tok % kS);
  int b = (int)(tok / kS);
  int r = spos[2 * s], c = spos[2 * s + 1];
  gridb[(((size_t)b * kG + r) * kG + c) * kD + d] = h[idx];
}

// ---------------- 3 dilated grouped 3x3 convs, fused combine + gather ----------------
// one block per (b, s): stage 7x7x256 patch in LDS, each thread = one out channel
__global__ __launch_bounds__(256) void conv_kernel(
    const float* __restrict__ gridb,
    const float* __restrict__ w1, const float* __restrict__ cb1,
    const float* __restrict__ w2, const float* __restrict__ cb2,
    const float* __restrict__ w3, const float* __restrict__ cb3,
    const int* __restrict__ spos,
    float* __restrict__ hout, __bf16* __restrict__ hbf) {
  __shared__ float patch[49 * kD];
  const int blk = blockIdx.x;
  const int b = blk / kS, s = blk - b * kS;
  const int t = threadIdx.x;
  const int cy = spos[2 * s], cx = spos[2 * s + 1];
  for (int p = 0; p < 49; ++p) {
    int y = cy + p / 7 - 3;
    int xx = cx + p % 7 - 3;
    float v = 0.f;
    if (y >= 0 && y < kG && xx >= 0 && xx < kG)
      v = gridb[(((size_t)b * kG + y) * kG + xx) * kD + t];
    patch[p * kD + t] = v;
  }
  __syncthreads();
  const int oc = t;
  const int gbase = (oc >> 5) << 5;   // input-channel group base
  const float* wp1 = w1 + (size_t)oc * 288;
  const float* wp2 = w2 + (size_t)oc * 288;
  const float* wp3 = w3 + (size_t)oc * 288;
  float a1 = 0.f, a2 = 0.f, a3 = 0.f;
#pragma unroll
  for (int ky = 0; ky < 3; ++ky) {
#pragma unroll
    for (int kx = 0; kx < 3; ++kx) {
      const int dy = ky - 1, dx = kx - 1;
      const float* p1 = patch + ((3 + dy) * 7 + 3 + dx) * kD + gbase;
      const float* p2 = patch + ((3 + 2 * dy) * 7 + 3 + 2 * dx) * kD + gbase;
      const float* p3 = patch + ((3 + 3 * dy) * 7 + 3 + 3 * dx) * kD + gbase;
      const int wo = ky * 3 + kx;
      for (int ic = 0; ic < 32; ++ic) {
        a1 = fmaf(wp1[ic * 9 + wo], p1[ic], a1);
        a2 = fmaf(wp2[ic * 9 + wo], p2[ic], a2);
        a3 = fmaf(wp3[ic * 9 + wo], p3[ic], a3);
      }
    }
  }
  float res = 0.5f * (patch[24 * kD + oc]
                      + fmaxf(a1 + cb1[oc], 0.f)
                      + fmaxf(a2 + cb2[oc], 0.f)
                      + fmaxf(a3 + cb3[oc], 0.f));
  const size_t i = (size_t)blk * kD + oc;
  hout[i] = res;
  hbf[i] = (__bf16)res;
}

// ---------------- host driver ----------------
extern "C" void kernel_launch(void* const* d_in, const int* in_sizes, int n_in,
                              void* d_out, int out_size, void* d_ws, size_t ws_size,
                              hipStream_t stream) {
  const int*   x         = (const int*)  d_in[0];
  const int*   spos      = (const int*)  d_in[1];
  const float* event_emb = (const float*)d_in[2];
  const float* pos_x     = (const float*)d_in[3];
  const float* pos_y     = (const float*)d_in[4];
  const float* stab_emb  = (const float*)d_in[5];
  const int*   tok_x     = (const int*)  d_in[6];
  const int*   tok_y     = (const int*)  d_in[7];
  const int*   tok_stab  = (const int*)  d_in[8];
  const float* attn_bias = (const float*)d_in[9];
  const float* Wqkv      = (const float*)d_in[10];
  const float* bqkv      = (const float*)d_in[11];
  const float* Wo        = (const float*)d_in[12];
  const float* bo        = (const float*)d_in[13];
  const float* W1        = (const float*)d_in[14];
  const float* b1        = (const float*)d_in[15];
  const float* W2        = (const float*)d_in[16];
  const float* b2        = (const float*)d_in[17];
  const float* ln1_s     = (const float*)d_in[18];
  const float* ln1_b     = (const float*)d_in[19];
  const float* ln2_s     = (const float*)d_in[20];
  const float* ln2_b     = (const float*)d_in[21];
  const float* cw1       = (const float*)d_in[22];
  const float* cb1       = (const float*)d_in[23];
  const float* cw2       = (const float*)d_in[24];
  const float* cb2       = (const float*)d_in[25];
  const float* cw3       = (const float*)d_in[26];
  const float* cb3       = (const float*)d_in[27];

  char* ws = (char*)d_ws;
  size_t off = 0;
  auto alloc = [&](size_t bytes) -> void* {
    void* p = ws + off;
    off = (off + bytes + 255) & ~(size_t)255;
    return p;
  };
  float*  h_f32  = (float*) alloc((size_t)kNTOK * kD * 4);
  __bf16* h_bf   = (__bf16*)alloc((size_t)kNTOK * kD * 2);
  __bf16* qkv_bf = (__bf16*)alloc((size_t)(kNTOK + 16) * 3 * kD * 2);
  float*  tmpF   = (float*) alloc((size_t)kNTOK * kD * 4);
  __bf16* o_bf   = (__bf16*)alloc((size_t)kNTOK * kD * 2);
  __bf16* ff_bf  = (__bf16*)alloc((size_t)kNTOK * kF * 2);
  float*  gridb  = (float*) alloc((size_t)kB * kG * kG * kD * 4);
  __bf16* wq_bf  = (__bf16*)alloc((size_t)kL * 3 * kD * kD * 2);
  __bf16* wo_bf  = (__bf16*)alloc((size_t)kL * kD * kD * 2);
  __bf16* w1_bf  = (__bf16*)alloc((size_t)kL * kF * kD * 2);
  __bf16* w2_bf  = (__bf16*)alloc((size_t)kL * kD * kF * 2);

  // weight conversion (f32 -> bf16)
  {
    long n;
    n = (long)kL * 3 * kD * kD;
    cvt_kernel<<<(n + 255) / 256, 256, 0, stream>>>(Wqkv, wq_bf, n);
    n = (long)kL * kD * kD;
    cvt_kernel<<<(n + 255) / 256, 256, 0, stream>>>(Wo, wo_bf, n);
    n = (long)kL * kF * kD;
    cvt_kernel<<<(n + 255) / 256, 256, 0, stream>>>(W1, w1_bf, n);
    n = (long)kL * kD * kF;
    cvt_kernel<<<(n + 255) / 256, 256, 0, stream>>>(W2, w2_bf, n);
  }

  // embedding
  embed_kernel<<<kNTOK, 256, 0, stream>>>(x, event_emb, pos_x, pos_y, stab_emb,
                                          tok_x, tok_y, tok_stab, h_f32, h_bf);

  const long gridN = (long)kB * kG * kG * kD;

  for (int l = 0; l < kL; ++l) {
    // QKV projection: (NTOK x 256) x (768 x 256)^T -> qkv_bf
    gemm_bf16_kernel<<<dim3(kMT32, 12), 32, 0, stream>>>(
        h_bf, wq_bf + (size_t)l * 3 * kD * kD, bqkv + (size_t)l * 3 * kD,
        nullptr, qkv_bf, 3 * kD, kD, 0);

    // fused attention
    attn_kernel<<<dim3(28, kB * kH), 32, 0, stream>>>(qkv_bf, attn_bias, o_bf);

    // output projection -> tmpF
    gemm_bf16_kernel<<<dim3(kMT32, 4), 32, 0, stream>>>(
        o_bf, wo_bf + (size_t)l * kD * kD, bo + (size_t)l * kD,
        tmpF, nullptr, kD, kD, 0);

    // h = LN(h + attn)
    ln_kernel<<<kNTOK, 256, 0, stream>>>(h_f32, tmpF, ln1_s + (size_t)l * kD,
                                         ln1_b + (size_t)l * kD, h_bf);

    // FFN1: relu(h W1^T + b1) -> ff_bf
    gemm_bf16_kernel<<<dim3(kMT32, 8), 32, 0, stream>>>(
        h_bf, w1_bf + (size_t)l * kF * kD, b1 + (size_t)l * kF,
        nullptr, ff_bf, kF, kD, 1);

    // FFN2: ff W2^T + b2 -> tmpF
    gemm_bf16_kernel<<<dim3(kMT32, 4), 32, 0, stream>>>(
        ff_bf, w2_bf + (size_t)l * kD * kF, b2 + (size_t)l * kD,
        tmpF, nullptr, kD, kF, 0);

    // h = LN(h + ff)  (bf16 mirror not needed before conv)
    ln_kernel<<<kNTOK, 256, 0, stream>>>(h_f32, tmpF, ln2_s + (size_t)l * kD,
                                         ln2_b + (size_t)l * kD, nullptr);

    // scatter to grid (zero first)
    zero_kernel<<<4096, 256, 0, stream>>>(gridb, gridN);
    scatter_kernel<<<kNTOK, 256, 0, stream>>>(h_f32, spos, gridb);

    // 3 dilated grouped convs + combine + gather -> new h (f32 + bf16)
    conv_kernel<<<kNTOK, 256, 0, stream>>>(
        gridb,
        cw1 + (size_t)l * kD * 32 * 9, cb1 + (size_t)l * kD,
        cw2 + (size_t)l * kD * 32 * 9, cb2 + (size_t)l * kD,
        cw3 + (size_t)l * kD * 32 * 9, cb3 + (size_t)l * kD,
        spos, h_f32, h_bf);
  }

  hipMemcpyAsync(d_out, h_f32, (size_t)kNTOK * kD * sizeof(float),
                 hipMemcpyDeviceToDevice, stream);
}
